// MambaModel_8718783611650
// MI455X (gfx1250) — matrix operand order repaired and checked by reading, not compiled, not run on hardware
//
#include <hip/hip_runtime.h>
#include <math.h>

// ---------------- problem constants ----------------
#define B_   128
#define C_   200
#define HW_  1024            // H*W = sequence length L
#define D_   256
#define S_   16
#define KC_  4               // conv taps
#define NL_  6
#define NC_  16
#define M_   (B_*HW_)        // 131072 rows
#define KE_  224             // embed K padded 200 -> 224 (multiple of 32)
#define EPS_ 1e-5f

typedef __attribute__((ext_vector_type(16))) __bf16 v16bf;
typedef __attribute__((ext_vector_type(8)))  float  v8f;

static __device__ __forceinline__ float  bf2f(__bf16 x){ return (float)x; }
static __device__ __forceinline__ __bf16 f2bf(float  x){ return (__bf16)x; }

// ---- CDNA5 async global->LDS path (guarded: falls back to sync copies) ----
#if defined(__HIP_DEVICE_COMPILE__) && \
    __has_builtin(__builtin_amdgcn_global_load_async_to_lds_b128) && \
    __has_builtin(__builtin_amdgcn_global_load_async_to_lds_b64) &&  \
    __has_builtin(__builtin_amdgcn_s_wait_asynccnt)
#define HAS_ASYNC_LDS 1
#else
#define HAS_ASYNC_LDS 0
#endif

#if HAS_ASYNC_LDS
// builtin params are (global int-vector*, lds int-vector*, imm offset, imm cpol)
typedef int v4i_vs __attribute__((vector_size(16)));
typedef int v2i_vs __attribute__((vector_size(8)));
typedef __attribute__((address_space(1))) void     gas_void;
typedef __attribute__((address_space(3))) void     lds_void;
typedef __attribute__((address_space(1))) v4i_vs*  gptr_b128;
typedef __attribute__((address_space(3))) v4i_vs*  lptr_b128;
typedef __attribute__((address_space(1))) v2i_vs*  gptr_b64;
typedef __attribute__((address_space(3))) v2i_vs*  lptr_b64;

static __device__ __forceinline__ void async_cp_b128(const __bf16* g, __bf16* l){
    __builtin_amdgcn_global_load_async_to_lds_b128(
        (gptr_b128)(gas_void*)(void*)g, (lptr_b128)(lds_void*)(void*)l, 0, 0);
}
static __device__ __forceinline__ void async_cp_b64(const __bf16* g, __bf16* l){
    __builtin_amdgcn_global_load_async_to_lds_b64(
        (gptr_b64)(gas_void*)(void*)g, (lptr_b64)(lds_void*)(void*)l, 0, 0);
}
#endif

// =====================================================================
// Generic bf16 WMMA GEMM:  C(M,N) = A(M,K) * B(N,K)^T   (both row-major,
// B stored as [N][K] so "out[n] = sum_k a[k]*w[n][k]" like `h @ W.T`)
// Block: 256 threads = 8 waves. Wave tile 32x32 (2x2 of 16x16x32 WMMA).
// Waves arranged WR (along M) x WC (along N); BM = WR*32, BN = WC*32.
// Double-buffered LDS; async global->LDS staging when available.
// =====================================================================
template<int BM, int BN, int WR, int WC, class Epi>
__global__ __launch_bounds__(256) void gemm_bf16_kernel(
    const __bf16* __restrict__ A, const __bf16* __restrict__ Bw,
    int K, Epi epi)
{
    constexpr int BK = 32;
    // async ops issued per *thread* per stage (uniform across all waves so the
    // s_wait_asynccnt immediate is exact): A uses b128, small-B uses b64.
    constexpr int  A_OPS   = (BM * BK) / (256 * 8);
    constexpr bool B_SMALL = (BN * BK) < (256 * 8);
    constexpr int  B_OPS   = B_SMALL ? 1 : (BN * BK) / (256 * 8);
    constexpr int  PER_STAGE = A_OPS + B_OPS;

    __shared__ alignas(32) __bf16 As[2][BM * BK];
    __shared__ alignas(32) __bf16 Bs[2][BN * BK];

    const int tid  = threadIdx.x;
    const int lane = tid & 31;
    const int wave = tid >> 5;
    const int laneRow = lane & 15;
    const int laneHi  = lane >> 4;
    const int wm = (wave % WR) * 32;
    const int wn = (wave / WR) * 32;
    const size_t m0 = (size_t)blockIdx.x * BM;
    const size_t n0 = (size_t)blockIdx.y * BN;

    v8f acc00 = {}, acc01 = {}, acc10 = {}, acc11 = {};
    union AFrag { v16bf v; uint4 u[2]; };

    auto stage = [&](int k0, int buf) {
#if HAS_ASYNC_LDS
#pragma unroll
        for (int i = 0; i < A_OPS; i++) {
            int f = tid * 8 + i * 2048;
            int r = f >> 5, c = f & 31;
            async_cp_b128(&A[(m0 + r) * (size_t)K + k0 + c], &As[buf][f]);
        }
        if (B_SMALL) {
            int f = tid * 4;                       // b64 = 4 bf16 per thread
            int r = f >> 5, c = f & 31;
            async_cp_b64(&Bw[(n0 + r) * (size_t)K + k0 + c], &Bs[buf][f]);
        } else {
#pragma unroll
            for (int i = 0; i < B_OPS; i++) {
                int f = tid * 8 + i * 2048;
                int r = f >> 5, c = f & 31;
                async_cp_b128(&Bw[(n0 + r) * (size_t)K + k0 + c], &Bs[buf][f]);
            }
        }
#else
        for (int f = tid * 8; f < BM * BK; f += 256 * 8) {
            int r = f >> 5, c = f & 31;
            *(uint4*)(&As[buf][f]) = *(const uint4*)(&A[(m0 + r) * (size_t)K + k0 + c]);
        }
        for (int f = tid * 8; f < BN * BK; f += 256 * 8) {
            int r = f >> 5, c = f & 31;
            *(uint4*)(&Bs[buf][f]) = *(const uint4*)(&Bw[(n0 + r) * (size_t)K + k0 + c]);
        }
#endif
    };

    const int nIter = K / BK;
    stage(0, 0);

    for (int it = 0; it < nIter; ++it) {
        const int buf = it & 1;
        if (it + 1 < nIter) stage((it + 1) * BK, buf ^ 1);
#if HAS_ASYNC_LDS
        if (it + 1 < nIter) __builtin_amdgcn_s_wait_asynccnt(PER_STAGE);
        else                __builtin_amdgcn_s_wait_asynccnt(0);
#endif
        __syncthreads();

        // A fragment: lane half g holds
        // row (wm+laneRow), K = [16g..16g+15] (contiguous, matches B)
        AFrag a0, a1;
        {
            const __bf16* r0 = &As[buf][(wm +      laneRow) * BK];
            const __bf16* r1 = &As[buf][(wm + 16 + laneRow) * BK];
            a0.u[0] = *(const uint4*)(r0 + 16 * laneHi);
            a0.u[1] = *(const uint4*)(r0 + 8 + 16 * laneHi);
            a1.u[0] = *(const uint4*)(r1 + 16 * laneHi);
            a1.u[1] = *(const uint4*)(r1 + 8 + 16 * laneHi);
        }
        // B fragment: lane holds column (wn..)+laneRow, 16 contiguous K
        v16bf b0 = *(const v16bf*)&Bs[buf][(wn +      laneRow) * BK + 16 * laneHi];
        v16bf b1 = *(const v16bf*)&Bs[buf][(wn + 16 + laneRow) * BK + 16 * laneHi];

        acc00 = __builtin_amdgcn_wmma_f32_16x16x32_bf16(false, a0.v, false, b0, (short)0, acc00, false, false);
        acc01 = __builtin_amdgcn_wmma_f32_16x16x32_bf16(false, a0.v, false, b1, (short)0, acc01, false, false);
        acc10 = __builtin_amdgcn_wmma_f32_16x16x32_bf16(false, a1.v, false, b0, (short)0, acc10, false, false);
        acc11 = __builtin_amdgcn_wmma_f32_16x16x32_bf16(false, a1.v, false, b1, (short)0, acc11, false, false);
        __syncthreads();
    }

    // Epilogue: C/D layout -> VGPR e: M = e + 8*laneHi, N = laneRow
#pragma unroll
    for (int i = 0; i < 2; i++)
#pragma unroll
        for (int j = 0; j < 2; j++) {
            const v8f a = (i == 0) ? ((j == 0) ? acc00 : acc01)
                                   : ((j == 0) ? acc10 : acc11);
#pragma unroll
            for (int e = 0; e < 8; e++) {
                size_t row = m0 + wm + 16 * i + e + 8 * laneHi;
                int    col = (int)n0 + wn + 16 * j + laneRow;
                epi(row, col, a[e]);
            }
        }
}

// ------------------------- epilogue functors -------------------------
struct EpiEmbed {                         // h = acc + bias  -> bf16
    const float* bias; __bf16* h;
    __device__ void operator()(size_t m, int n, float v) const {
        h[m * D_ + n] = f2bf(v + bias[n]);
    }
};
struct EpiSplit {                         // inp proj: split u_pre / res
    const float* bias; __bf16* u; __bf16* r;
    __device__ void operator()(size_t m, int n, float v) const {
        v += bias[n];
        if (n < D_) u[m * D_ + n] = f2bf(v);
        else        r[m * D_ + (n - D_)] = f2bf(v);
    }
};
struct EpiDADB {                          // cols 0..15 -> exp -> dA, 16..31 -> dB
    float* dA; float* dB;
    __device__ void operator()(size_t m, int n, float v) const {
        if (n < S_) dA[m * S_ + n] = __expf(v);
        else        dB[m * S_ + (n - S_)] = v;
    }
};
struct EpiYSilu {                         // g = (y @ Cw.T) * silu(res)
    const __bf16* res; __bf16* g;
    __device__ void operator()(size_t m, int n, float v) const {
        float r = bf2f(res[m * D_ + n]);
        float s = r / (1.f + __expf(-r));
        g[m * D_ + n] = f2bf(v * s);
    }
};
struct EpiOut {                           // o = acc + bias + u
    const float* bias; const __bf16* u; float* o;
    __device__ void operator()(size_t m, int n, float v) const {
        o[m * D_ + n] = v + bias[n] + bf2f(u[m * D_ + n]);
    }
};

// =====================================================================
// Support kernels
// =====================================================================

// per-channel batchnorm stats over (B, HW) for x[B,C,HW]
__global__ void bnx_stats_kernel(const float* __restrict__ x,
                                 float* __restrict__ meanC,
                                 float* __restrict__ rstdC)
{
    int c = blockIdx.x, t = threadIdx.x;
    float s = 0.f, q = 0.f;
    for (int b = 0; b < B_; b++) {
        const float* p = x + ((size_t)b * C_ + c) * HW_;
        for (int j = t; j < HW_; j += 256) { float v = p[j]; s += v; q += v * v; }
    }
    __shared__ float rs[256], rq[256];
    rs[t] = s; rq[t] = q; __syncthreads();
    for (int o = 128; o > 0; o >>= 1) {
        if (t < o) { rs[t] += rs[t + o]; rq[t] += rq[t + o]; }
        __syncthreads();
    }
    if (t == 0) {
        float inv = 1.f / (float)(B_ * HW_);
        float mu = rs[0] * inv;
        float var = rq[0] * inv - mu * mu;
        meanC[c] = mu; rstdC[c] = rsqrtf(var + EPS_);
    }
}

// build bf16 embed A matrix (M x 224): normalized + NHWC transpose + K-pad
__global__ void prep_embed_A_kernel(const float* __restrict__ x,
                                    const float* __restrict__ meanC,
                                    const float* __restrict__ rstdC,
                                    __bf16* __restrict__ A)
{
    int m = blockIdx.x * 256 + threadIdx.x;   // grid = M_/256
    int b = m >> 10, hw = m & 1023;
    for (int c = 0; c < KE_; c++) {
        float v = 0.f;
        if (c < C_)
            v = (x[((size_t)b * C_ + c) * HW_ + hw] - meanC[c]) * rstdC[c];
        A[(size_t)m * KE_ + c] = f2bf(v);
    }
}

// fp32 -> bf16 weight convert with column pad and optional transpose
__global__ void convert_pad_kernel(const float* __restrict__ src,
                                   __bf16* __restrict__ dst,
                                   int dcols, int valid, int src_ld,
                                   int transpose, int total)
{
    int idx = blockIdx.x * 256 + threadIdx.x;
    if (idx >= total) return;
    int r = idx / dcols, c = idx % dcols;
    float v = 0.f;
    if (c < valid)
        v = transpose ? src[(size_t)c * src_ld + r] : src[(size_t)r * src_ld + c];
    dst[idx] = f2bf(v);
}

// causal depthwise conv along L (K=4 taps), bf16 in/out, fp32 math
__global__ void dwconv_kernel(const __bf16* __restrict__ up,
                              const float* __restrict__ w,
                              const float* __restrict__ cb,
                              __bf16* __restrict__ u)
{
    size_t idx = (size_t)blockIdx.x * 256 + threadIdx.x;   // over M_*D_
    int d = (int)(idx & (D_ - 1));
    size_t m = idx >> 8;
    int l = (int)(m & (HW_ - 1));
    float acc = cb[d];
#pragma unroll
    for (int k = 0; k < KC_; k++) {
        int off = (KC_ - 1) - k;              // o[l] += w[k]*u[l-(K-1)+k]
        if (l - off >= 0) acc += w[d * KC_ + k] * bf2f(up[(m - off) * D_ + d]);
    }
    u[idx] = f2bf(acc);
}

// reference scan: cp=cumprod(dA); y=cumsum(cp*dB); out[l]=y[l]-y[l-1]*dA[l]
// one block per batch row, 16 lanes = 16 states; writes K-padded bf16 y
__global__ void scan_kernel(const float* __restrict__ dA,
                            const float* __restrict__ dB,
                            __bf16* __restrict__ yP)
{
    int b = blockIdx.x, s = threadIdx.x;      // blockDim = 16
    float cp = 1.f, cum = 0.f;
    for (int l = 0; l < HW_; l++) {
        size_t id = ((size_t)b * HW_ + l) * S_ + s;
        float a  = dA[id];
        float bv = dB[id];
        cp *= a;
        float cumn = cum + cp * bv;
        float outv = (l == 0) ? bv : (cumn - cum * a);
        cum = cumn;
        size_t yo = ((size_t)b * HW_ + l) * 32;
        yP[yo + s]      = f2bf(outv);
        yP[yo + 16 + s] = f2bf(0.f);          // K pad
    }
}

// per-feature column stats (coalesced rows + atomic partials)
__global__ void bn_col_partial_kernel(const float* __restrict__ o,
                                      float* __restrict__ gsum,
                                      float* __restrict__ gsq)
{
    int t = threadIdx.x;                      // t = feature (D_ = 256)
    size_t m0 = (size_t)blockIdx.x * 256;
    float s = 0.f, q = 0.f;
    for (int r = 0; r < 256; r++) {
        float v = o[(m0 + r) * D_ + t];
        s += v; q += v * v;
    }
    atomicAdd(&gsum[t], s);
    atomicAdd(&gsq[t], q);
}

__global__ void bn_finalize_kernel(const float* gsum, const float* gsq,
                                   float* mean, float* rstd, float invN)
{
    int t = threadIdx.x;
    float mu = gsum[t] * invN;
    float var = gsq[t] * invN - mu * mu;
    mean[t] = mu; rstd[t] = rsqrtf(var + EPS_);
}

// normalize o in place (fp32) and emit bf16 h for the next layer's GEMM
__global__ void bn_apply_kernel(float* __restrict__ o, __bf16* __restrict__ h,
                                const float* __restrict__ mean,
                                const float* __restrict__ rstd)
{
    size_t idx = (size_t)blockIdx.x * 256 + threadIdx.x;  // over M_*D_
    int d = (int)(idx & (D_ - 1));
    float v = (o[idx] - mean[d]) * rstd[d];
    o[idx] = v;
    h[idx] = f2bf(v);
}

// final layernorm over D per (b,l), then mean over l -> hmean[b][d]
__global__ void final_ln_mean_kernel(const float* __restrict__ o,
                                     const float* __restrict__ g,
                                     const float* __restrict__ beta,
                                     float* __restrict__ hmean)
{
    int b = blockIdx.x, t = threadIdx.x;      // 256 threads = D features
    __shared__ float rs[256], rq[256];
    __shared__ float smu, srstd;
    float acc = 0.f;
    for (int l = 0; l < HW_; l++) {
        float v = o[((size_t)b * HW_ + l) * D_ + t];
        rs[t] = v; rq[t] = v * v; __syncthreads();
        for (int ofs = 128; ofs > 0; ofs >>= 1) {
            if (t < ofs) { rs[t] += rs[t + ofs]; rq[t] += rq[t + ofs]; }
            __syncthreads();
        }
        if (t == 0) {
            float mu = rs[0] * (1.f / D_);
            float var = rq[0] * (1.f / D_) - mu * mu;
            smu = mu; srstd = rsqrtf(var + EPS_);
        }
        __syncthreads();
        acc += (v - smu) * srstd * g[t] + beta[t];
        __syncthreads();
    }
    hmean[(size_t)b * D_ + t] = acc * (1.f / HW_);
}

// batch-BN over b for hmean(128,256), then head GEMM (128x16x256), tiny
__global__ void final_head_kernel(float* __restrict__ hmean,
                                  const float* __restrict__ hW,
                                  const float* __restrict__ hb,
                                  float* __restrict__ out)
{
    int t = threadIdx.x;                      // 256 threads, single block
    float s = 0.f, q = 0.f;
    for (int b = 0; b < B_; b++) {
        float v = hmean[(size_t)b * D_ + t]; s += v; q += v * v;
    }
    float mu = s * (1.f / B_);
    float var = q * (1.f / B_) - mu * mu;
    float rst = rsqrtf(var + EPS_);
    for (int b = 0; b < B_; b++)
        hmean[(size_t)b * D_ + t] = (hmean[(size_t)b * D_ + t] - mu) * rst;
    __syncthreads();
    for (int idx = t; idx < B_ * NC_; idx += 256) {
        int b = idx >> 4, n = idx & 15;
        float acc = hb[n];
        for (int d = 0; d < D_; d++)
            acc += hmean[(size_t)b * D_ + d] * hW[n * D_ + d];
        out[idx] = acc;
    }
}

// =====================================================================
// Host driver
// =====================================================================
extern "C" void kernel_launch(void* const* d_in, const int* in_sizes, int n_in,
                              void* d_out, int out_size, void* d_ws, size_t ws_size,
                              hipStream_t stream)
{
    const float* x       = (const float*)d_in[0];
    const float* embed_W = (const float*)d_in[1];
    const float* embed_b = (const float*)d_in[2];
    const float* dt      = (const float*)d_in[3];
    const float* Bw      = (const float*)d_in[4];
    const float* Cw      = (const float*)d_in[5];
    const float* conv_w  = (const float*)d_in[6];
    const float* conv_b  = (const float*)d_in[7];
    const float* inp_W   = (const float*)d_in[8];
    const float* inp_b   = (const float*)d_in[9];
    const float* out_W   = (const float*)d_in[10];
    const float* out_b   = (const float*)d_in[11];
    const float* ln_g    = (const float*)d_in[12];
    const float* ln_bb   = (const float*)d_in[13];
    const float* head_W  = (const float*)d_in[14];
    const float* head_b  = (const float*)d_in[15];
    float* out = (float*)d_out;
    (void)in_sizes; (void)n_in; (void)out_size; (void)ws_size;

    char* ws = (char*)d_ws;
    size_t off = 0;
    auto alloc = [&](size_t bytes) -> char* {
        char* p = ws + off;
        off = (off + bytes + 255) & ~(size_t)255;
        return p;
    };

    float*  meanC = (float*)alloc(256 * 4);
    float*  rstdC = (float*)alloc(256 * 4);
    float*  gsum  = (float*)alloc(2 * 256 * 4);        // gsum + gsq contiguous
    float*  gsq   = gsum + 256;
    float*  meanD = (float*)alloc(256 * 4);
    float*  rstdD = (float*)alloc(256 * 4);
    __bf16* A0    = (__bf16*)alloc((size_t)M_ * KE_ * 2);
    __bf16* eWb   = (__bf16*)alloc((size_t)D_ * KE_ * 2);
    __bf16* iWb   = (__bf16*)alloc((size_t)NL_ * 2 * D_ * D_ * 2);
    __bf16* oWb   = (__bf16*)alloc((size_t)NL_ * D_ * D_ * 2);
    __bf16* dtBwb = (__bf16*)alloc((size_t)NL_ * 32 * D_ * 2);
    __bf16* CwPb  = (__bf16*)alloc((size_t)NL_ * D_ * 32 * 2);
    __bf16* h_bf  = (__bf16*)alloc((size_t)M_ * D_ * 2);
    __bf16* upre  = (__bf16*)alloc((size_t)M_ * D_ * 2);
    __bf16* res   = (__bf16*)alloc((size_t)M_ * D_ * 2);
    __bf16* u_bf  = (__bf16*)alloc((size_t)M_ * D_ * 2);
    float*  dA    = (float*)alloc((size_t)M_ * S_ * 4);
    float*  dB    = (float*)alloc((size_t)M_ * S_ * 4);
    __bf16* yP    = (__bf16*)alloc((size_t)M_ * 32 * 2);
    __bf16* g_bf  = (__bf16*)alloc((size_t)M_ * D_ * 2);
    float*  o_f   = (float*)alloc((size_t)M_ * D_ * 4);
    float*  hmean = (float*)alloc((size_t)B_ * D_ * 4);

    // ---- input batchnorm + embed A prep ----
    bnx_stats_kernel<<<C_, 256, 0, stream>>>(x, meanC, rstdC);
    prep_embed_A_kernel<<<M_ / 256, 256, 0, stream>>>(x, meanC, rstdC, A0);

    // ---- weight conversion to bf16 (padded / transposed as needed) ----
    convert_pad_kernel<<<(D_ * KE_ + 255) / 256, 256, 0, stream>>>(
        embed_W, eWb, KE_, C_, C_, 0, D_ * KE_);
    for (int i = 0; i < NL_; i++) {
        convert_pad_kernel<<<(2 * D_ * D_ + 255) / 256, 256, 0, stream>>>(
            inp_W + (size_t)i * 2 * D_ * D_, iWb + (size_t)i * 2 * D_ * D_,
            D_, D_, D_, 0, 2 * D_ * D_);
        convert_pad_kernel<<<(D_ * D_ + 255) / 256, 256, 0, stream>>>(
            out_W + (size_t)i * D_ * D_, oWb + (size_t)i * D_ * D_,
            D_, D_, D_, 0, D_ * D_);
        // dt (D,S) transposed into rows 0..15 of the stacked [dtT;Bw] B-matrix
        convert_pad_kernel<<<(S_ * D_ + 255) / 256, 256, 0, stream>>>(
            dt + (size_t)i * D_ * S_, dtBwb + (size_t)i * 32 * D_,
            D_, D_, S_, 1, S_ * D_);
        // Bw (S,D) direct into rows 16..31
        convert_pad_kernel<<<(S_ * D_ + 255) / 256, 256, 0, stream>>>(
            Bw + (size_t)i * S_ * D_, dtBwb + (size_t)i * 32 * D_ + S_ * D_,
            D_, D_, D_, 0, S_ * D_);
        // Cw (D,S) padded to (D,32)
        convert_pad_kernel<<<(D_ * 32 + 255) / 256, 256, 0, stream>>>(
            Cw + (size_t)i * D_ * S_, CwPb + (size_t)i * D_ * 32,
            32, S_, S_, 0, D_ * 32);
    }

    // ---- embed GEMM: h = A0(M,224) @ eWb(256,224)^T + b ----
    gemm_bf16_kernel<128, 64, 4, 2><<<dim3(M_ / 128, D_ / 64), 256, 0, stream>>>(
        A0, eWb, KE_, EpiEmbed{embed_b, h_bf});

    // ---- layers ----
    for (int i = 0; i < NL_; i++) {
        const __bf16* iW_i = iWb  + (size_t)i * 2 * D_ * D_;
        const __bf16* oW_i = oWb  + (size_t)i * D_ * D_;
        const __bf16* sB_i = dtBwb + (size_t)i * 32 * D_;
        const __bf16* Cw_i = CwPb + (size_t)i * D_ * 32;

        // in-projection: (M,256)x(512,256)^T -> split u_pre / res
        gemm_bf16_kernel<128, 64, 4, 2><<<dim3(M_ / 128, (2 * D_) / 64), 256, 0, stream>>>(
            h_bf, iW_i, D_, EpiSplit{inp_b + (size_t)i * 2 * D_, upre, res});

        // causal depthwise conv
        dwconv_kernel<<<(M_ * D_) / 256, 256, 0, stream>>>(
            upre, conv_w + (size_t)i * D_ * KC_, conv_b + (size_t)i * D_, u_bf);

        // dA = exp(u@dt), dB = u@Bw^T via stacked N=32 GEMM
        gemm_bf16_kernel<256, 32, 8, 1><<<dim3(M_ / 256, 1), 256, 0, stream>>>(
            u_bf, sB_i, D_, EpiDADB{dA, dB});

        // sequential selective scan (reference recurrence), K-padded bf16 out
        scan_kernel<<<B_, 16, 0, stream>>>(dA, dB, yP);

        // y @ Cw^T, gated by silu(res)
        gemm_bf16_kernel<128, 64, 4, 2><<<dim3(M_ / 128, D_ / 64), 256, 0, stream>>>(
            yP, Cw_i, 32, EpiYSilu{res, g_bf});

        // out-projection + bias + residual u
        gemm_bf16_kernel<128, 64, 4, 2><<<dim3(M_ / 128, D_ / 64), 256, 0, stream>>>(
            g_bf, oW_i, D_, EpiOut{out_b + (size_t)i * D_, u_bf, o_f});

        // batchnorm over (B, L) per feature
        (void)hipMemsetAsync(gsum, 0, 2 * 256 * sizeof(float), stream);
        bn_col_partial_kernel<<<M_ / 256, 256, 0, stream>>>(o_f, gsum, gsq);
        bn_finalize_kernel<<<1, 256, 0, stream>>>(gsum, gsq, meanD, rstdD, 1.f / (float)M_);
        bn_apply_kernel<<<(M_ * D_) / 256, 256, 0, stream>>>(o_f, h_bf, meanD, rstdD);
    }

    // ---- final layernorm + mean over L, batch-BN + head ----
    final_ln_mean_kernel<<<B_, 256, 0, stream>>>(o_f, ln_g, ln_bb, hmean);
    final_head_kernel<<<1, 256, 0, stream>>>(hmean, head_W, head_b, out);
}
